// PredictionNet_75608604279246
// MI455X (gfx1250) — matrix operand order repaired and checked by reading, not compiled, not run on hardware
//
#include <hip/hip_runtime.h>

// ---------------------------------------------------------------------------
// Types for CDNA5 WMMA
// ---------------------------------------------------------------------------
typedef __attribute__((ext_vector_type(16))) __bf16 v16bf;
typedef __attribute__((ext_vector_type(8)))  float  v8f;

union Frag { uint4 u[2]; unsigned int w[8]; v16bf v; };

__device__ __forceinline__ __bf16 bf16_of_f32(float f) {
  unsigned int x = __builtin_bit_cast(unsigned int, f);
  unsigned short r = (unsigned short)((x + 0x7FFFu + ((x >> 16) & 1u)) >> 16);
  return __builtin_bit_cast(__bf16, r);
}

// Async global->LDS copy, 16 bytes/lane, GVS form: scalar 64-bit base +
// 32-bit per-lane offset. LDS dest VGPR = wave-relative LDS byte offset.
__device__ __forceinline__ void async_ld128_gvs(unsigned lds, unsigned voff,
                                                unsigned long long sbase) {
  asm volatile("global_load_async_to_lds_b128 %0, %1, %2"
               :: "v"(lds), "v"(voff), "s"(sbase) : "memory");
}
__device__ __forceinline__ void wait_async0() {
  asm volatile("s_wait_asynccnt 0x0" ::: "memory");
}

// ---------------------------------------------------------------------------
// Prep kernel 1: h0 = concat(frame_prev[B,1830], I[B,256]) -> bf16 [B, 2112]
// ---------------------------------------------------------------------------
__global__ __launch_bounds__(256) void pack_h0_kernel(
    const float* __restrict__ fp, const float* __restrict__ ii,
    __bf16* __restrict__ out) {
  long idx = (long)blockIdx.x * 256 + threadIdx.x;   // B*2112 threads
  int b = (int)(idx / 2112);
  int k = (int)(idx % 2112);
  float v = 0.0f;
  if (k < 1830)      v = fp[(long)b * 1830 + k];
  else if (k < 2086) v = ii[(long)b * 256 + (k - 1830)];
  out[idx] = bf16_of_f32(v);
}

// ---------------------------------------------------------------------------
// Prep kernel 2: wp[e][n][kp] = bf16(w[e][k][n]), zero-padded to KP
// ---------------------------------------------------------------------------
__global__ __launch_bounds__(256) void pack_w_kernel(
    const float* __restrict__ w, __bf16* __restrict__ out, int K, int KP) {
  long idx = (long)blockIdx.x * 256 + threadIdx.x;   // 6*512*KP threads
  int kp = (int)(idx % KP);
  int n  = (int)((idx / KP) % 512);
  int e  = (int)(idx / ((long)KP * 512));
  float v = (kp < K) ? w[((long)e * K + kp) * 512 + n] : 0.0f;
  out[idx] = bf16_of_f32(v);
}

// ---------------------------------------------------------------------------
// Blended-expert GEMM + bias + ELU.
//   sum[m,n] = sum_e coef[m,e] * (sum_k A[m,k]*W[e,n,k])   (fold per expert)
//   out      = elu(sum + sum_e coef[m,e]*bias[e,n])
// Block: 256 threads = 8 waves; tile 128(M) x 64(N) x 64(K-stage);
// per wave 32x32 (2x2 WMMA frags, 2 K sub-steps per stage).
// Tiles move global->LDS via GLOBAL_LOAD_ASYNC_TO_LDS_B128 (ASYNCcnt),
// double-buffered, 1 barrier + 1 asynccnt-wait per stage.
// ---------------------------------------------------------------------------
template <int KP, bool LAST>
__global__ __launch_bounds__(256) void moe_gemm_kernel(
    const __bf16* __restrict__ A,     // [B][KP] bf16, row-major
    const __bf16* __restrict__ W,     // [6][512][KP] bf16 (N-major, K contig)
    const float*  __restrict__ coef,  // [B][6]
    const float*  __restrict__ bias,  // [6][512]
    __bf16* __restrict__ outb,        // [B][512] bf16 (next layer), or null
    float*  __restrict__ outf)        // [B][512] f32 (final), or null
{
  constexpr int KST = 64;            // K elements per LDS stage
  constexpr int KCH = KP / KST;      // stages per expert (33 or 8)
  constexpr int NIT = 6 * KCH;       // total stages

  __shared__ __align__(16) __bf16 sA[2][128 * KST];   // 2 x 16 KB
  __shared__ __align__(16) __bf16 sB[2][64 * KST];    // 2 x  8 KB
  __shared__ float sCoef[128 * 6];                    //      3 KB

  const int t    = threadIdx.x;
  const int lane = t & 31;
  const int wave = t >> 5;       // 0..7
  const int wm   = wave >> 1;    // 0..3  -> 32-row band
  const int wn   = wave & 1;     // 0..1  -> 32-col band
  const int bm   = blockIdx.x;   // 0..63
  const int bn   = blockIdx.y;   // 0..7
  const int l15  = lane & 15;
  const int lhi  = lane >> 4;    // 0/1

  // stage per-row expert coefficients for this 128-row block
  for (int i = t; i < 128 * 6; i += 256)
    sCoef[i] = coef[(long)(bm * 128 + i / 6) * 6 + (i % 6)];

  const v8f vzero = (v8f){0.f, 0.f, 0.f, 0.f, 0.f, 0.f, 0.f, 0.f};
  v8f acc[2][2], sum[2][2];
#pragma unroll
  for (int i = 0; i < 2; ++i)
#pragma unroll
    for (int j = 0; j < 2; ++j) { acc[i][j] = vzero; sum[i][j] = vzero; }

  // ---- loop-invariant async-copy addressing --------------------------------
  // Tile element id u = t + i*256 (uint4 units). row = u/8, col8 = u%8.
  // Per-lane global byte offset (same formula for A rows 0..127 and B rows
  // 0..63): ((t>>3) + i*32)*KP*2 + (t&7)*16. Scalar base carries bm/e/kc.
  unsigned voff[4];
#pragma unroll
  for (int i = 0; i < 4; ++i)
    voff[i] = (unsigned)(((t >> 3) + i * 32) * (KP * 2) + (t & 7) * 16);

  unsigned ldsA[2][4], ldsB[2][2];
#pragma unroll
  for (int b = 0; b < 2; ++b) {
#pragma unroll
    for (int i = 0; i < 4; ++i)
      ldsA[b][i] = (unsigned)(uintptr_t)&sA[b][0] + t * 16 + i * 4096;
#pragma unroll
    for (int i = 0; i < 2; ++i)
      ldsB[b][i] = (unsigned)(uintptr_t)&sB[b][0] + t * 16 + i * 4096;
  }

  // scalar (uniform) base pointers for the NEXT stage to issue
  unsigned long long aNext = (unsigned long long)(uintptr_t)A
                             + (unsigned long long)bm * 128 * KP * 2;
  unsigned long long bNext = (unsigned long long)(uintptr_t)W
                             + (unsigned long long)bn * 64 * KP * 2;
  int kcIss = 0;

  auto issue = [&](int buf) {
#pragma unroll
    for (int i = 0; i < 4; ++i) async_ld128_gvs(ldsA[buf][i], voff[i], aNext);
#pragma unroll
    for (int i = 0; i < 2; ++i) async_ld128_gvs(ldsB[buf][i], voff[i], bNext);
    if (++kcIss == KCH) {   // advance to next expert (uniform SALU math)
      kcIss = 0;
      aNext -= (unsigned long long)(KCH - 1) * (KST * 2);
      bNext += 1024ull * KP - (unsigned long long)(KCH - 1) * (KST * 2);
    } else {
      aNext += KST * 2;
      bNext += KST * 2;
    }
  };

  issue(0);

  int kcc = 0, efold = 0;
  for (int it = 0; it < NIT; ++it) {
    const int buf = it & 1;

    wait_async0();        // my async loads for `it` have landed in LDS[buf]
    __syncthreads();      // everyone's loads landed; everyone done with buf^1
    if (it + 1 < NIT) issue((it + 1) & 1);   // overlap next stage with compute

    // two K=32 sub-steps over the 64-deep stage
#pragma unroll
    for (int s = 0; s < 2; ++s) {
      const int ks = s * 32;
      // A fragments: lane row = l15; lanes 0-15 K=0..7,16..23; lanes 16-31 +8
      Frag fa[2];
      const int kq = ks + lhi * 8;
#pragma unroll
      for (int fm = 0; fm < 2; ++fm) {
        const int row = wm * 32 + fm * 16 + l15;
        fa[fm].u[0] = *(const uint4*)&sA[buf][row * KST + kq];
        fa[fm].u[1] = *(const uint4*)&sA[buf][row * KST + 16 + kq];
      }
      // B fragments: lane col = l15; lanes 0-15 K=0..7,16..23; lanes 16-31 +8
      Frag fb[2];
      const int kb = ks + lhi * 8;
#pragma unroll
      for (int fn = 0; fn < 2; ++fn) {
        const int col = wn * 32 + fn * 16 + l15;
        fb[fn].u[0] = *(const uint4*)&sB[buf][col * KST + kb];
        fb[fn].u[1] = *(const uint4*)&sB[buf][col * KST + kb + 16];
      }
#pragma unroll
      for (int fm = 0; fm < 2; ++fm)
#pragma unroll
        for (int fn = 0; fn < 2; ++fn)
          acc[fm][fn] = __builtin_amdgcn_wmma_f32_16x16x32_bf16(
              false, fa[fm].v, false, fb[fn].v, (short)0, acc[fm][fn],
              false, false);
    }

    // expert boundary: fold acc into sum with fp32 coef, reset acc
    if (++kcc == KCH) {
      kcc = 0;
#pragma unroll
      for (int fm = 0; fm < 2; ++fm) {
#pragma unroll
        for (int r = 0; r < 8; ++r) {
          const float c = sCoef[(wm * 32 + fm * 16 + r + 8 * lhi) * 6 + efold];
#pragma unroll
          for (int fn = 0; fn < 2; ++fn)
            sum[fm][fn][r] += c * acc[fm][fn][r];
        }
      }
#pragma unroll
      for (int fm = 0; fm < 2; ++fm)
#pragma unroll
        for (int fn = 0; fn < 2; ++fn)
          acc[fm][fn] = vzero;
      ++efold;
    }
  }

  // ---- epilogue: + sum_e coef*bias, ELU, store -----------------------------
#pragma unroll
  for (int fn = 0; fn < 2; ++fn) {
    const int n = bn * 64 + wn * 32 + fn * 16 + l15;   // global col
    float be[6];
#pragma unroll
    for (int e = 0; e < 6; ++e) be[e] = bias[e * 512 + n];
#pragma unroll
    for (int fm = 0; fm < 2; ++fm) {
#pragma unroll
      for (int r = 0; r < 8; ++r) {
        const int mloc = wm * 32 + fm * 16 + r + 8 * lhi;
        float x = sum[fm][fn][r];
#pragma unroll
        for (int e = 0; e < 6; ++e) x += sCoef[mloc * 6 + e] * be[e];
        x = (x > 0.0f) ? x : (__expf(x) - 1.0f);       // ELU(alpha=1)
        const long m = (long)bm * 128 + mloc;
        if (LAST) outf[m * 512 + n] = x;
        else      outb[m * 512 + n] = bf16_of_f32(x);
      }
    }
  }
}

// ---------------------------------------------------------------------------
// launch
// ---------------------------------------------------------------------------
extern "C" void kernel_launch(void* const* d_in, const int* in_sizes, int n_in,
                              void* d_out, int out_size, void* d_ws, size_t ws_size,
                              hipStream_t stream) {
  const float* frame_prev = (const float*)d_in[0];
  const float* I_in       = (const float*)d_in[1];
  const float* coef       = (const float*)d_in[2];
  const float* w1         = (const float*)d_in[3];
  const float* b1         = (const float*)d_in[4];
  const float* w2         = (const float*)d_in[5];
  const float* b2         = (const float*)d_in[6];
  const float* w3         = (const float*)d_in[7];
  const float* b3         = (const float*)d_in[8];
  float* out = (float*)d_out;

  char* ws = (char*)d_ws;
  __bf16* h0  = (__bf16*)(ws);                       // 8192*2112*2 = 34,603,008
  __bf16* w1p = (__bf16*)(ws + 34603008);            // 6*512*2112*2 = 12,976,128
  __bf16* w2p = (__bf16*)(ws + 47579136);            // 6*512*512*2 =  3,145,728
  __bf16* w3p = (__bf16*)(ws + 50724864);            //                3,145,728
  __bf16* h1  = (__bf16*)(ws + 53870592);            // 8192*512*2  =  8,388,608
  __bf16* h2  = (__bf16*)(ws + 62259200);            //                8,388,608

  pack_h0_kernel<<<(8192L * 2112) / 256, 256, 0, stream>>>(frame_prev, I_in, h0);
  pack_w_kernel<<<(6L * 512 * 2112) / 256, 256, 0, stream>>>(w1, w1p, 2086, 2112);
  pack_w_kernel<<<(6L * 512 * 512) / 256, 256, 0, stream>>>(w2, w2p, 512, 512);
  pack_w_kernel<<<(6L * 512 * 512) / 256, 256, 0, stream>>>(w3, w3p, 512, 512);

  dim3 grid(8192 / 128, 512 / 64);   // 64 x 8
  moe_gemm_kernel<2112, false><<<grid, 256, 0, stream>>>(h0, w1p, coef, b1, h1, nullptr);
  moe_gemm_kernel<512,  false><<<grid, 256, 0, stream>>>(h1, w2p, coef, b2, h2, nullptr);
  moe_gemm_kernel<512,  true ><<<grid, 256, 0, stream>>>(h2, w3p, coef, b3, nullptr, out);
}
